// Diffusion_6794638262642
// MI455X (gfx1250) — matrix operand order repaired and checked by reading, not compiled, not run on hardware
//
#include <hip/hip_runtime.h>

// ---------------------------------------------------------------------------
// CDNA5 (gfx1250): wave32, WMMA bf16->f32, double-buffered LDS tiles fed by
// async global->LDS DMA (ASYNCcnt).  Weights stored pre-transposed so both
// GEMM operands are K-major -> no in-loop transposes, tile fills are pure
// global_load_async_to_lds_b128.
// ---------------------------------------------------------------------------

typedef __attribute__((ext_vector_type(16))) __bf16 v16bf;
typedef __attribute__((ext_vector_type(8)))  float  v8f;
typedef __attribute__((ext_vector_type(4)))  int    v4i;

#define Dv    1024
#define Bv    128
#define Nv    576
#define BNv   (Bv * Nv)      // 73728
#define D2v   2048

#define ACT_NONE 0
#define ACT_SILU 1

// ----- async global->LDS (gfx1250) with safe fallback ----------------------
#if defined(__AMDGCN__) && __has_builtin(__builtin_amdgcn_global_load_async_to_lds_b128)
#define HAVE_ASYNC 1
typedef __attribute__((address_space(1))) v4i as1_v4i;   // global
typedef __attribute__((address_space(3))) v4i as3_v4i;   // LDS
#else
#define HAVE_ASYNC 0
#endif

__device__ __forceinline__ void copy16_g2l(const __bf16* g, __bf16* l) {
#if HAVE_ASYNC
    __builtin_amdgcn_global_load_async_to_lds_b128((as1_v4i*)g, (as3_v4i*)l, 0, 0);
#else
    *(uint4*)l = *(const uint4*)g;
#endif
}

__device__ __forceinline__ void wait_g2l() {
#if HAVE_ASYNC
#if __has_builtin(__builtin_amdgcn_s_wait_asynccnt)
    __builtin_amdgcn_s_wait_asynccnt(0);
#else
    asm volatile("s_wait_asynccnt 0" ::: "memory");
#endif
#endif
}

// ---------------------------------------------------------------------------
// 32x32 LDS-tiled transpose + fp32->bf16:  dst[c][r] = bf16(src[r][c])
// Coalesced on both the load and the store side.
// ---------------------------------------------------------------------------
__global__ __launch_bounds__(256) void transconv_kernel(const float* __restrict__ src,
                                                        __bf16* __restrict__ dst,
                                                        int rows, int cols) {
    __shared__ float tile[32][33];
    const int c0 = blockIdx.x * 32, r0 = blockIdx.y * 32;
    const int lr  = threadIdx.x >> 3;        // 0..31
    const int lc4 = (threadIdx.x & 7) * 4;   // 0,4,...,28
    float4 v = *(const float4*)&src[(size_t)(r0 + lr) * cols + c0 + lc4];
    tile[lr][lc4 + 0] = v.x; tile[lr][lc4 + 1] = v.y;
    tile[lr][lc4 + 2] = v.z; tile[lr][lc4 + 3] = v.w;
    __syncthreads();
    union { __bf16 h[4]; uint2 u; } pk;
    pk.h[0] = (__bf16)tile[lc4 + 0][lr]; pk.h[1] = (__bf16)tile[lc4 + 1][lr];
    pk.h[2] = (__bf16)tile[lc4 + 2][lr]; pk.h[3] = (__bf16)tile[lc4 + 3][lr];
    *(uint2*)&dst[(size_t)(c0 + lr) * rows + r0 + lc4] = pk.u;
}

// vectorized fp32 -> bf16 (8 elems / thread)
__global__ __launch_bounds__(256) void convert8_kernel(const float* __restrict__ src,
                                                       __bf16* __restrict__ dst) {
    int idx = blockIdx.x * 256 + threadIdx.x;
    const float4 a = *(const float4*)&src[idx * 8];
    const float4 b = *(const float4*)&src[idx * 8 + 4];
    union { __bf16 h[8]; uint4 u; } pk;
    pk.h[0] = (__bf16)a.x; pk.h[1] = (__bf16)a.y; pk.h[2] = (__bf16)a.z; pk.h[3] = (__bf16)a.w;
    pk.h[4] = (__bf16)b.x; pk.h[5] = (__bf16)b.y; pk.h[6] = (__bf16)b.z; pk.h[7] = (__bf16)b.w;
    *(uint4*)&dst[idx * 8] = pk.u;
}

// ---------------------------------------------------------------------------
// sinusoidal timestep embedding -> bf16 [B, D]
// ---------------------------------------------------------------------------
__global__ __launch_bounds__(256) void temb_kernel(const int* __restrict__ ts,
                                                   __bf16* __restrict__ out) {
    int idx = blockIdx.x * 256 + threadIdx.x;
    int b = idx >> 10, d = idx & 1023;
    float t   = (float)ts[b];
    float arg = t * __expf((float)(d & ~1) * (-logf(10000.0f) / (float)Dv));
    float v   = (d & 1) ? __cosf(arg) : __sinf(arg);
    out[idx] = (__bf16)v;
}

// aqb = bf16(text_feat + cond)
__global__ __launch_bounds__(256) void addc_kernel(const float* __restrict__ a,
                                                   const float* __restrict__ b,
                                                   __bf16* __restrict__ out) {
    int idx = blockIdx.x * 256 + threadIdx.x;
    out[idx] = (__bf16)(a[idx] + b[idx]);
}

// ---------------------------------------------------------------------------
// bf16 WMMA GEMM: C[128, N] = act(A[128,K] @ W + bias), with W given
// TRANSPOSED (BT[n][k], row stride ldbt).  Block tile 128x64, 8 waves,
// double-buffered LDS, all tile fills via async global->LDS b128 copies.
// ---------------------------------------------------------------------------
template <int ACT, bool BIAS, bool OF, bool OB>
__global__ __launch_bounds__(256) void gemm_bf16(
    const __bf16* __restrict__ A, const __bf16* __restrict__ BT,
    const float* __restrict__ bias,
    float* __restrict__ outF, __bf16* __restrict__ outB,
    int K, int N, int ldbt) {

    __shared__ __align__(16) __bf16 sA[2][128 * 32];    // [row][k]
    __shared__ __align__(16) __bf16 sBT[2][64 * 32];    // [n][k]

    const int t    = threadIdx.x;
    const int w    = t >> 5;
    const int lane = t & 31;
    const int half = lane >> 4;
    const int ln   = lane & 15;
    const int n0   = blockIdx.x * 64;
    const int nk   = K / 32;

    auto loadA = [&](int buf, int k0) {
        #pragma unroll
        for (int i = 0; i < 2; ++i) {
            int idx = t + 256 * i;
            int r = idx >> 2, c4 = idx & 3;
            copy16_g2l(&A[r * K + k0 + c4 * 8], &sA[buf][r * 32 + c4 * 8]);
        }
    };
    auto loadB = [&](int buf, int k0) {
        int n = t >> 2, kq = t & 3;
        copy16_g2l(&BT[(size_t)(n0 + n) * ldbt + k0 + kq * 8], &sBT[buf][n * 32 + kq * 8]);
    };

    v8f acc[4] = {};
    loadA(0, 0); loadB(0, 0);
    wait_g2l(); __syncthreads();

    for (int kt = 0; kt < nk; ++kt) {
        const int cur = kt & 1;
        if (kt + 1 < nk) { loadA(cur ^ 1, (kt + 1) * 32); loadB(cur ^ 1, (kt + 1) * 32); }

        const __bf16* sa = sA[cur];
        const __bf16* sb = sBT[cur];
        v16bf af;
        {
            uint4* ap = (uint4*)&af;
            int m = 16 * w + ln;
            ap[0] = *(const uint4*)&sa[m * 32 + 8 * half];
            ap[1] = *(const uint4*)&sa[m * 32 + 16 + 8 * half];
        }
        v16bf bf4[4];
        #pragma unroll
        for (int s = 0; s < 4; ++s) {
            uint4* bp_ = (uint4*)&bf4[s];
            int nn = s * 16 + ln;
            bp_[0] = *(const uint4*)&sb[nn * 32 + 8 * half];
            bp_[1] = *(const uint4*)&sb[nn * 32 + 16 + 8 * half];
        }
        #pragma unroll
        for (int s = 0; s < 4; ++s)
            acc[s] = __builtin_amdgcn_wmma_f32_16x16x32_bf16(
                false, af, false, bf4[s], (short)0, acc[s], false, false);

        wait_g2l();
        __syncthreads();
    }

    #pragma unroll
    for (int s = 0; s < 4; ++s) {
        int col = n0 + s * 16 + ln;
        float bv = 0.0f;
        if constexpr (BIAS) bv = bias[col];
        #pragma unroll
        for (int g = 0; g < 8; ++g) {
            int row = 16 * w + g + 8 * half;
            float v = acc[s][g] + bv;
            if constexpr (ACT == ACT_SILU) v = v * (1.0f / (1.0f + __expf(-v)));
            if constexpr (OF) outF[row * N + col] = v;
            if constexpr (OB) outB[row * N + col] = (__bf16)v;
        }
    }
}

// ---------------------------------------------------------------------------
// s[b,n] = dot(text[b], image[b,n])
// ---------------------------------------------------------------------------
__global__ __launch_bounds__(256) void sim_kernel(const float* __restrict__ text,
                                                  const float* __restrict__ img,
                                                  float* __restrict__ s) {
    int wv = threadIdx.x >> 5, lane = threadIdx.x & 31;
    int bn = blockIdx.x * 8 + wv;
    int b  = bn / Nv;
    const float* tf = text + b * Dv;
    const float* im = img + (size_t)bn * Dv;
    float acc = 0.0f;
    #pragma unroll
    for (int i = 0; i < 8; ++i) {
        int d = i * 128 + lane * 4;
        float4 a = *(const float4*)&tf[d];
        float4 c = *(const float4*)&im[d];
        acc += a.x * c.x + a.y * c.y + a.z * c.z + a.w * c.w;
    }
    #pragma unroll
    for (int off = 16; off >= 1; off >>= 1) acc += __shfl_xor(acc, off);
    if (lane == 0) s[bn] = acc;   // TEMP == 1
}

// logits[b,n] = dot(image[b,n]*iw[b,n] + cond[b], qk[b]) + x[b,n]
__global__ __launch_bounds__(256) void logits_kernel(const float* __restrict__ img,
                                                     const float* __restrict__ iw,
                                                     const float* __restrict__ cond,
                                                     const float* __restrict__ qk,
                                                     const float* __restrict__ x,
                                                     float* __restrict__ out) {
    int wv = threadIdx.x >> 5, lane = threadIdx.x & 31;
    int bn = blockIdx.x * 8 + wv;
    int b  = bn / Nv;
    float iwv = iw[bn];
    const float* im = img + (size_t)bn * Dv;
    const float* cd = cond + b * Dv;
    const float* qq = qk + b * Dv;
    float acc = 0.0f;
    #pragma unroll
    for (int i = 0; i < 8; ++i) {
        int d = i * 128 + lane * 4;
        float4 a = *(const float4*)&im[d];
        float4 c = *(const float4*)&cd[d];
        float4 q = *(const float4*)&qq[d];
        acc += (a.x * iwv + c.x) * q.x + (a.y * iwv + c.y) * q.y
             + (a.z * iwv + c.z) * q.z + (a.w * iwv + c.w) * q.w;
    }
    #pragma unroll
    for (int off = 16; off >= 1; off >>= 1) acc += __shfl_xor(acc, off);
    if (lane == 0) out[bn] = acc + x[bn];
}

// row softmax over `cols`
__global__ __launch_bounds__(256) void softmax_rows(const float* __restrict__ in,
                                                    float* __restrict__ out, int cols) {
    __shared__ float red[256];
    int b = blockIdx.x, t = threadIdx.x;
    const float* r = in + b * cols;
    float m = -1e30f;
    for (int c = t; c < cols; c += 256) m = fmaxf(m, r[c]);
    red[t] = m; __syncthreads();
    for (int s = 128; s > 0; s >>= 1) { if (t < s) red[t] = fmaxf(red[t], red[t + s]); __syncthreads(); }
    m = red[0]; __syncthreads();
    float ss = 0.0f;
    for (int c = t; c < cols; c += 256) ss += __expf(r[c] - m);
    red[t] = ss; __syncthreads();
    for (int s = 128; s > 0; s >>= 1) { if (t < s) red[t] += red[t + s]; __syncthreads(); }
    float inv = 1.0f / red[0];
    for (int c = t; c < cols; c += 256) out[b * cols + c] = __expf(r[c] - m) * inv;
}

// abarb[b,:] = bf16( sum_n weight*iw*image[b,n,:] + cond[b,:] )
__global__ __launch_bounds__(256) void abar_kernel(const float* __restrict__ img,
                                                   const float* __restrict__ wgt,
                                                   const float* __restrict__ iw,
                                                   const float* __restrict__ cond,
                                                   __bf16* __restrict__ out) {
    int b = blockIdx.x, t = threadIdx.x;
    float acc[4] = {0.f, 0.f, 0.f, 0.f};
    for (int n = 0; n < Nv; ++n) {
        int bn = b * Nv + n;
        float wn = wgt[bn] * iw[bn];
        const float* row = img + (size_t)bn * Dv;
        #pragma unroll
        for (int i = 0; i < 4; ++i) acc[i] += wn * row[t + 256 * i];
    }
    #pragma unroll
    for (int i = 0; i < 4; ++i)
        out[b * Dv + t + 256 * i] = (__bf16)(acc[i] + cond[b * Dv + t + 256 * i]);
}

// ---------------------------------------------------------------------------
// Fused decoder:  out[bn] = bd2 + weight[bn]
//   + sum_j relu( iw[bn]*(imgb[bn] @ Wd1bot)[j] + c1[b,j] ) * Wd2[j]
// Wd1botT = transposed bottom half of Wd1 (row stride 2048).
// ---------------------------------------------------------------------------
__global__ __launch_bounds__(256) void decoder_kernel(
    const __bf16* __restrict__ imgb, const float* __restrict__ iw,
    const float* __restrict__ c1, const __bf16* __restrict__ Wd1botT,
    const float* __restrict__ Wd2, const float* __restrict__ bd2,
    const float* __restrict__ wgt, float* __restrict__ out) {

    __shared__ __align__(16) __bf16 sA[2][128 * 32];
    __shared__ __align__(16) __bf16 sBT[2][64 * 32];

    const int t    = threadIdx.x;
    const int w    = t >> 5;
    const int lane = t & 31;
    const int half = lane >> 4;
    const int ln   = lane & 15;
    const int row0 = blockIdx.x * 128;
    const __bf16* Ab = imgb + (size_t)row0 * Dv;

    float pacc[8];
    float iwg[8];
    int   bofs[8];
    #pragma unroll
    for (int g = 0; g < 8; ++g) {
        int row = row0 + 16 * w + g + 8 * half;
        pacc[g] = 0.0f;
        iwg[g]  = iw[row];
        bofs[g] = (row / Nv) * D2v;
    }

    auto loadA = [&](int buf, int k0) {
        #pragma unroll
        for (int i = 0; i < 2; ++i) {
            int idx = t + 256 * i;
            int r = idx >> 2, c4 = idx & 3;
            copy16_g2l(&Ab[r * Dv + k0 + c4 * 8], &sA[buf][r * 32 + c4 * 8]);
        }
    };
    auto loadB = [&](int buf, int k0, int chunk) {
        int n = t >> 2, kq = t & 3;
        copy16_g2l(&Wd1botT[(size_t)(chunk + n) * D2v + k0 + kq * 8],
                   &sBT[buf][n * 32 + kq * 8]);
    };

    for (int chunk = 0; chunk < D2v; chunk += 64) {
        v8f acc[4] = {};
        loadA(0, 0); loadB(0, 0, chunk);
        wait_g2l(); __syncthreads();

        for (int kt = 0; kt < Dv / 32; ++kt) {
            const int cur = kt & 1;
            if (kt + 1 < Dv / 32) { loadA(cur ^ 1, (kt + 1) * 32); loadB(cur ^ 1, (kt + 1) * 32, chunk); }

            const __bf16* sa = sA[cur];
            const __bf16* sb = sBT[cur];
            v16bf af;
            {
                uint4* ap = (uint4*)&af;
                int m = 16 * w + ln;
                ap[0] = *(const uint4*)&sa[m * 32 + 8 * half];
                ap[1] = *(const uint4*)&sa[m * 32 + 16 + 8 * half];
            }
            v16bf bf4[4];
            #pragma unroll
            for (int s = 0; s < 4; ++s) {
                uint4* bp_ = (uint4*)&bf4[s];
                int nn = s * 16 + ln;
                bp_[0] = *(const uint4*)&sb[nn * 32 + 8 * half];
                bp_[1] = *(const uint4*)&sb[nn * 32 + 16 + 8 * half];
            }
            #pragma unroll
            for (int s = 0; s < 4; ++s)
                acc[s] = __builtin_amdgcn_wmma_f32_16x16x32_bf16(
                    false, af, false, bf4[s], (short)0, acc[s], false, false);

            wait_g2l();
            __syncthreads();
        }

        // fused epilogue: *iw, +c1, relu, *Wd2; sum across the 4 subtiles
        // first, then a single 16-lane reduction per row accumulator.
        float wc[4];
        #pragma unroll
        for (int s = 0; s < 4; ++s) wc[s] = Wd2[chunk + s * 16 + ln];
        #pragma unroll
        for (int g = 0; g < 8; ++g) {
            float v = 0.0f;
            #pragma unroll
            for (int s = 0; s < 4; ++s) {
                int col = chunk + s * 16 + ln;
                float u = acc[s][g] * iwg[g] + c1[bofs[g] + col];
                v += fmaxf(u, 0.0f) * wc[s];
            }
            v += __shfl_xor(v, 8);
            v += __shfl_xor(v, 4);
            v += __shfl_xor(v, 2);
            v += __shfl_xor(v, 1);
            pacc[g] += v;
        }
    }

    if (ln == 0) {
        float b2 = bd2[0];
        #pragma unroll
        for (int g = 0; g < 8; ++g) {
            int row = row0 + 16 * w + g + 8 * half;
            out[row] = pacc[g] + b2 + wgt[row];
        }
    }
}

// ---------------------------------------------------------------------------
// Host orchestration
// ---------------------------------------------------------------------------
extern "C" void kernel_launch(void* const* d_in, const int* in_sizes, int n_in,
                              void* d_out, int out_size, void* d_ws, size_t ws_size,
                              hipStream_t stream) {
    const float* x     = (const float*)d_in[0];
    const int*   ts    = (const int*)  d_in[1];
    const float* text  = (const float*)d_in[2];
    const float* imgf  = (const float*)d_in[3];
    const float* Wq    = (const float*)d_in[4];
    const float* Wk    = (const float*)d_in[5];
    const float* Wv    = (const float*)d_in[6];
    const float* Wp    = (const float*)d_in[7];
    const float* bp    = (const float*)d_in[8];
    const float* Wt1   = (const float*)d_in[9];
    const float* bt1   = (const float*)d_in[10];
    const float* Wt2   = (const float*)d_in[11];
    const float* bt2   = (const float*)d_in[12];
    const float* Wd1   = (const float*)d_in[13];
    const float* bd1   = (const float*)d_in[14];
    const float* Wd2   = (const float*)d_in[15];
    const float* bd2   = (const float*)d_in[16];

    char*  ws  = (char*)d_ws;
    size_t off = 0;
    auto alloc = [&](size_t bytes) -> void* {
        void* p = (void*)(ws + off);
        off += (bytes + 255) & ~(size_t)255;
        return p;
    };

    const size_t DD = (size_t)Dv * Dv;
    __bf16* Wt1T  = (__bf16*)alloc(DD * 2);              // transposed bf16 weights
    __bf16* Wt2T  = (__bf16*)alloc(DD * 2);
    __bf16* WqT   = (__bf16*)alloc(DD * 2);
    __bf16* Wkb   = (__bf16*)alloc(DD * 2);              // (Wk^T)^T == Wk, straight
    __bf16* WvT   = (__bf16*)alloc(DD * 2);
    __bf16* WpT   = (__bf16*)alloc(DD * 2);
    __bf16* Wd1T  = (__bf16*)alloc((size_t)D2v * D2v * 2);
    __bf16* imgb  = (__bf16*)alloc((size_t)BNv * Dv * 2);
    __bf16* tembb = (__bf16*)alloc((size_t)Bv * Dv * 2);
    __bf16* h1b   = (__bf16*)alloc((size_t)Bv * Dv * 2);
    float*  cond  = (float*) alloc((size_t)Bv * Dv * 4);
    __bf16* aqb   = (__bf16*)alloc((size_t)Bv * Dv * 2);
    __bf16* qb    = (__bf16*)alloc((size_t)Bv * Dv * 2);
    float*  qkf   = (float*) alloc((size_t)Bv * Dv * 4);
    float*  sbuf  = (float*) alloc((size_t)BNv * 4);
    float*  iwbuf = (float*) alloc((size_t)BNv * 4);
    float*  lbuf  = (float*) alloc((size_t)BNv * 4);
    float*  wbuf  = (float*) alloc((size_t)BNv * 4);
    __bf16* abarb = (__bf16*)alloc((size_t)Bv * Dv * 2);
    __bf16* tmpb  = (__bf16*)alloc((size_t)Bv * Dv * 2);
    __bf16* nembb = (__bf16*)alloc((size_t)Bv * Dv * 2);
    float*  c1    = (float*) alloc((size_t)Bv * D2v * 4);

    // 1. one-time precision conversion; weights stored transposed (K-major)
    transconv_kernel<<<dim3(Dv / 32, Dv / 32), 256, 0, stream>>>(Wt1, Wt1T, Dv, Dv);
    transconv_kernel<<<dim3(Dv / 32, Dv / 32), 256, 0, stream>>>(Wt2, Wt2T, Dv, Dv);
    transconv_kernel<<<dim3(Dv / 32, Dv / 32), 256, 0, stream>>>(Wq,  WqT,  Dv, Dv);
    transconv_kernel<<<dim3(Dv / 32, Dv / 32), 256, 0, stream>>>(Wv,  WvT,  Dv, Dv);
    transconv_kernel<<<dim3(Dv / 32, Dv / 32), 256, 0, stream>>>(Wp,  WpT,  Dv, Dv);
    transconv_kernel<<<dim3(D2v / 32, D2v / 32), 256, 0, stream>>>(Wd1, Wd1T, D2v, D2v);
    convert8_kernel<<<(int)(DD / 8 / 256), 256, 0, stream>>>(Wk, Wkb);
    convert8_kernel<<<(int)((size_t)BNv * Dv / 8 / 256), 256, 0, stream>>>(imgf, imgb);

    // 2. timestep embedding + conditioning MLP
    temb_kernel<<<Bv * Dv / 256, 256, 0, stream>>>(ts, tembb);
    gemm_bf16<ACT_SILU, true, false, true><<<Dv / 64, 256, 0, stream>>>(
        tembb, Wt1T, bt1, nullptr, h1b, Dv, Dv, Dv);
    gemm_bf16<ACT_NONE, true, true, false><<<Dv / 64, 256, 0, stream>>>(
        h1b, Wt2T, bt2, cond, nullptr, Dv, Dv, Dv);

    // 3. q = (text+cond)@Wq ;  qk = q @ Wk^T  (transposed-B buffer == Wk)
    addc_kernel<<<Bv * Dv / 256, 256, 0, stream>>>(text, cond, aqb);
    gemm_bf16<ACT_NONE, false, false, true><<<Dv / 64, 256, 0, stream>>>(
        aqb, WqT, nullptr, nullptr, qb, Dv, Dv, Dv);
    gemm_bf16<ACT_NONE, false, true, false><<<Dv / 64, 256, 0, stream>>>(
        qb, Wkb, nullptr, qkf, nullptr, Dv, Dv, Dv);

    // 4. image re-weighting softmax
    sim_kernel<<<BNv / 8, 256, 0, stream>>>(text, imgf, sbuf);
    softmax_rows<<<Bv, 256, 0, stream>>>(sbuf, iwbuf, Nv);

    // 5. attention logits + softmax
    logits_kernel<<<BNv / 8, 256, 0, stream>>>(imgf, iwbuf, cond, qkf, x, lbuf);
    softmax_rows<<<Bv, 256, 0, stream>>>(lbuf, wbuf, Nv);

    // 6. new_emb = (sum_n w*(img+cond)) @ Wv @ Wp + bp ;
    //    c1 = new_emb @ Wd1_top + bd1  (top half = Wd1T cols 0..1023)
    abar_kernel<<<Bv, 256, 0, stream>>>(imgf, wbuf, iwbuf, cond, abarb);
    gemm_bf16<ACT_NONE, false, false, true><<<Dv / 64, 256, 0, stream>>>(
        abarb, WvT, nullptr, nullptr, tmpb, Dv, Dv, Dv);
    gemm_bf16<ACT_NONE, true, false, true><<<Dv / 64, 256, 0, stream>>>(
        tmpb, WpT, bp, nullptr, nembb, Dv, Dv, Dv);
    gemm_bf16<ACT_NONE, true, true, false><<<D2v / 64, 256, 0, stream>>>(
        nembb, Wd1T, bd1, c1, nullptr, Dv, D2v, D2v);

    // 7. fused decoder GEMM (imgb @ Wd1_bot, *iw, relu, dot Wd2) + weight
    //    bottom half = Wd1T cols 1024..2047
    decoder_kernel<<<BNv / 128, 256, 0, stream>>>(imgb, iwbuf, c1,
                                                  Wd1T + Dv, Wd2, bd2,
                                                  wbuf, (float*)d_out);
    (void)in_sizes; (void)n_in; (void)out_size; (void)ws_size;
}
